// Net_17188459119114
// MI455X (gfx1250) — compile-verified, matrix operand order checked by reading, not run on hardware
//
#include <hip/hip_runtime.h>
#include <hip/hip_bf16.h>

typedef __attribute__((ext_vector_type(16))) _Float16 v16h;
typedef __attribute__((ext_vector_type(8)))  float    v8f;

#define B_IMG   4096
#define IH      14
#define IW      38
#define OC      32
#define OH      12
#define OW      36
#define PH      3
#define PW      9
#define KFEAT   864            // OC*PH*PW = 32*27
#define KCHUNKS 27             // 864 / 32
#define MTILES  (B_IMG / 16)   // 256

// ---------------------------------------------------------------------------
// Kernel A: binarize fc_w and pre-swizzle into WMMA B-fragment layout.
// B is 32x16 (KxN) per chunk, 16-bit: lane&15 = column n; lanes 0-15 hold
// K {0..7,16..23} in slots {0..7,8..15}; lanes 16-31 hold K {8..15,24..31}.
// One wave fills all 27 chunks: bbuf[(kc*32 + lane)*16 + slot].
// ---------------------------------------------------------------------------
__global__ void bnn_prep_b(const float* __restrict__ fc_w,
                           _Float16* __restrict__ bbuf) {
    const int lane = threadIdx.x;          // 0..31
    const int n    = lane & 15;
    const int hi   = lane >> 4;            // 0 or 1
    for (int kc = 0; kc < KCHUNKS; ++kc) {
        #pragma unroll
        for (int slot = 0; slot < 16; ++slot) {
            int ko = (slot & 7) + 8 * hi + 16 * (slot >> 3);
            int k  = kc * 32 + ko;
            float v = 0.f;
            if (n < 10) {
                float w = fc_w[n * KFEAT + k];
                v = (w > 0.f) ? 1.f : ((w < 0.f) ? -1.f : 0.f);   // jnp.sign
            }
            bbuf[(kc * 32 + lane) * 16 + slot] = (_Float16)v;
        }
    }
}

// ---------------------------------------------------------------------------
// Kernel B: fused BinaryConv3x3 + BatchNorm + ternary hardtanh + MaxPool(4).
// One block per image; image staged in LDS; each thread produces pooled
// outputs and writes them as f16 in the WMMA A-fragment swizzled layout:
// act[((tile*27 + kc)*32 + lane)*16 + slot], tile=b>>4, row m=b&15,
// lane = m + 16*((ko>>3)&1), slot = (ko&7) + 8*(ko>>4), ko = k&31.
// ---------------------------------------------------------------------------
__global__ __launch_bounds__(288)
void bnn_conv_pool(const float* __restrict__ x,
                   const float* __restrict__ conv_w,
                   const float* __restrict__ conv_b,
                   const float* __restrict__ bn_gamma,
                   const float* __restrict__ bn_beta,
                   const float* __restrict__ bn_mean,
                   const float* __restrict__ bn_var,
                   _Float16* __restrict__ act) {
    __shared__ float simg[IH * IW];        // 532 floats
    __shared__ float ssw[OC * 9];          // binarized conv weights
    __shared__ float sscale[OC], sshift[OC];

    const int b   = blockIdx.x;
    const int tid = threadIdx.x;

    // stage the input image
    for (int i = tid; i < IH * IW; i += blockDim.x)
        simg[i] = x[b * (IH * IW) + i];

    // per-channel constants (first 32 threads)
    if (tid < OC) {
        float inv = bn_gamma[tid] * rsqrtf(bn_var[tid] + 1e-5f);
        sscale[tid] = inv;
        sshift[tid] = (conv_b[tid] - bn_mean[tid]) * inv + bn_beta[tid];
        #pragma unroll
        for (int j = 0; j < 9; ++j) {
            float w = conv_w[tid * 9 + j];
            ssw[tid * 9 + j] = (w > 0.f) ? 1.f : ((w < 0.f) ? -1.f : 0.f);
        }
    }
    __syncthreads();

    const int tile = b >> 4;
    const int m    = b & 15;

    // 864 pooled outputs; k == idx == oc*27 + ph*9 + pw
    for (int idx = tid; idx < KFEAT; idx += blockDim.x) {
        const int oc = idx / 27;
        const int r  = idx - oc * 27;
        const int ph = r / 9;
        const int pw = r - ph * 9;

        // 6x6 input patch into registers (read LDS once)
        float p[6][6];
        #pragma unroll
        for (int i = 0; i < 6; ++i)
            #pragma unroll
            for (int j = 0; j < 6; ++j)
                p[i][j] = simg[(ph * 4 + i) * IW + (pw * 4 + j)];

        const float s  = sscale[oc];
        const float d  = sshift[oc];
        const float* w = &ssw[oc * 9];

        float mx = -2.f;
        #pragma unroll
        for (int dy = 0; dy < 4; ++dy) {
            #pragma unroll
            for (int dx = 0; dx < 4; ++dx) {
                float acc = 0.f;
                #pragma unroll
                for (int i = 0; i < 3; ++i)
                    #pragma unroll
                    for (int j = 0; j < 3; ++j)
                        acc = fmaf(p[dy + i][dx + j], w[i * 3 + j], acc);
                float v = fmaf(acc, s, d);
                // round-half-even of clip(v,-1,1): thresholds at +/-0.5 -> 0
                float t = (v > 0.5f) ? 1.f : ((v < -0.5f) ? -1.f : 0.f);
                mx = fmaxf(mx, t);
            }
        }

        const int kc   = idx >> 5;
        const int ko   = idx & 31;
        const int lane = m + 16 * ((ko >> 3) & 1);
        const int slot = (ko & 7) + 8 * (ko >> 4);
        act[(((size_t)tile * KCHUNKS + kc) * 32 + lane) * 16 + slot] =
            (_Float16)mx;
    }
}

// ---------------------------------------------------------------------------
// Kernel C: FC via WMMA. 4 waves/block, one 16x16 output tile per wave.
// K = 864 -> 27 chained v_wmma_f32_16x16x32_f16, unrolled 3x so the loads
// of 3 K-chunks are all issued before their 3 chained WMMAs.
// Exact arithmetic since all A/B values are in {-1,0,1}.
// ---------------------------------------------------------------------------
__global__ __launch_bounds__(128)
void bnn_fc_wmma(const _Float16* __restrict__ act,
                 const _Float16* __restrict__ bbuf,
                 const float* __restrict__ fc_b,
                 float* __restrict__ out) {
    const int wave = threadIdx.x >> 5;
    const int lane = threadIdx.x & 31;
    const int tile = blockIdx.x * 4 + wave;    // 0..255

    const _Float16* aptr =
        act + (((size_t)tile * KCHUNKS) * 32 + lane) * 16;
    const _Float16* bptr = bbuf + (size_t)lane * 16;

    v8f c = {};
    #pragma unroll 1
    for (int kc = 0; kc < KCHUNKS; kc += 3) {
        // issue all 6 loads (3 chunks of A and B) up front, then chain WMMAs
        v16h a0 = *(const v16h*)(aptr + (size_t)(kc + 0) * 32 * 16);
        v16h a1 = *(const v16h*)(aptr + (size_t)(kc + 1) * 32 * 16);
        v16h a2 = *(const v16h*)(aptr + (size_t)(kc + 2) * 32 * 16);
        v16h b0 = *(const v16h*)(bptr + (size_t)(kc + 0) * 32 * 16);
        v16h b1 = *(const v16h*)(bptr + (size_t)(kc + 1) * 32 * 16);
        v16h b2 = *(const v16h*)(bptr + (size_t)(kc + 2) * 32 * 16);
        c = __builtin_amdgcn_wmma_f32_16x16x32_f16(
                false, a0, false, b0, (short)0, c, false, false);
        c = __builtin_amdgcn_wmma_f32_16x16x32_f16(
                false, a1, false, b1, (short)0, c, false, false);
        c = __builtin_amdgcn_wmma_f32_16x16x32_f16(
                false, a2, false, b2, (short)0, c, false, false);
    }

    const int n = lane & 15;
    if (n < 10) {
        const float bias = fc_b[n];
        const int mbase  = 8 * (lane >> 4);
        #pragma unroll
        for (int j = 0; j < 8; ++j) {
            int mrow = mbase + j;
            out[(size_t)(tile * 16 + mrow) * 10 + n] = c[j] + bias;
        }
    }
}

// ---------------------------------------------------------------------------
extern "C" void kernel_launch(void* const* d_in, const int* in_sizes, int n_in,
                              void* d_out, int out_size, void* d_ws, size_t ws_size,
                              hipStream_t stream) {
    const float* x        = (const float*)d_in[0];
    const float* conv_w   = (const float*)d_in[1];
    const float* conv_b   = (const float*)d_in[2];
    const float* bn_gamma = (const float*)d_in[3];
    const float* bn_beta  = (const float*)d_in[4];
    const float* bn_mean  = (const float*)d_in[5];
    const float* bn_var   = (const float*)d_in[6];
    const float* fc_w     = (const float*)d_in[7];
    const float* fc_b     = (const float*)d_in[8];
    float* out            = (float*)d_out;

    // workspace: [0, 27648) = B fragments; [32768, +7.08MB) = activations
    _Float16* bbuf = (_Float16*)d_ws;
    _Float16* act  = (_Float16*)((char*)d_ws + 32768);

    bnn_prep_b<<<1, 32, 0, stream>>>(fc_w, bbuf);
    bnn_conv_pool<<<B_IMG, 288, 0, stream>>>(x, conv_w, conv_b, bn_gamma,
                                             bn_beta, bn_mean, bn_var, act);
    bnn_fc_wmma<<<MTILES / 4, 128, 0, stream>>>(act, bbuf, fc_b, out);
}
